// GTN_89137751261502
// MI455X (gfx1250) — matrix operand order repaired and checked by reading, not compiled, not run on hardware
//
#include <hip/hip_runtime.h>
#include <hip/hip_fp16.h>

// ---- problem constants (match reference) ----
#define NNODE 50000
#define NEDGE 600000
#define FDIM  128
#define NCLS  10

typedef __attribute__((ext_vector_type(16))) _Float16 v16h;
typedef __attribute__((ext_vector_type(8)))  _Float16 v8h;
typedef __attribute__((ext_vector_type(8)))  float    v8f;

// order-preserving float->uint encoding for atomicMax-based segment_max
static __device__ __forceinline__ unsigned enc_f32(float f) {
    unsigned u = __float_as_uint(f);
    return (u & 0x80000000u) ? ~u : (u | 0x80000000u);
}
static __device__ __forceinline__ float dec_f32(unsigned u) {
    u = (u & 0x80000000u) ? (u & 0x7FFFFFFFu) : ~u;
    return __uint_as_float(u);
}

// ---------------- elementwise helpers ----------------
__global__ void k_f32_to_f16(const float* __restrict__ src,
                             _Float16* __restrict__ dst, int n) {
    int i = blockIdx.x * blockDim.x + threadIdx.x;
    if (i < n) dst[i] = (_Float16)src[i];
}

__global__ void k_pack_weights(const float* __restrict__ Wq, const float* __restrict__ Wk,
                               const float* __restrict__ Wv, const float* __restrict__ Ws,
                               _Float16* __restrict__ w16, int layerOff) {
    int i = blockIdx.x * blockDim.x + threadIdx.x;          // 0 .. 4*128*128
    if (i >= 4 * FDIM * FDIM) return;
    int mat = i >> 14, off = i & 16383;
    const float* p = (mat == 0) ? Wq : (mat == 1) ? Wk : (mat == 2) ? Wv : Ws;
    w16[i] = (_Float16)p[layerOff + off];
}

__global__ void k_init_softmax(unsigned* __restrict__ menc, float* __restrict__ denom, int n) {
    int i = blockIdx.x * blockDim.x + threadIdx.x;
    if (i < n) { menc[i] = 0u; denom[i] = 0.f; }
}

__global__ void k_relu_to_h16(const float* __restrict__ agg,
                              _Float16* __restrict__ h16, int n) {
    int i = blockIdx.x * blockDim.x + threadIdx.x;
    if (i < n) h16[i] = (_Float16)fmaxf(agg[i], 0.f);
}

// ---------------- fused q/k/v/skip GEMM via WMMA ----------------
// grid = (N/16, 4); block = 256 (8 waves). blockIdx.y selects matrix
// (0=q,1=k,2=v,3=skip->agg). Each wave computes a 16x16 tile of
// out = h @ W^T + b, K-loop over 4 blocks of K=32 with
// v_wmma_f32_16x16x32_f16, f32 accumulation.
__global__ __launch_bounds__(256) void k_gemm_qkvs(
    const _Float16* __restrict__ h16, const _Float16* __restrict__ w16,
    const float* __restrict__ bq, const float* __restrict__ bk,
    const float* __restrict__ bv, const float* __restrict__ bs,
    float* __restrict__ oq, float* __restrict__ okk,
    float* __restrict__ ov, float* __restrict__ os) {
    const int lane   = threadIdx.x & 31;
    const int wave   = threadIdx.x >> 5;    // 0..7 -> column tile
    const int laneLo = lane & 15;
    const int laneHi = lane >> 4;           // 0 or 1
    const int mat    = blockIdx.y;
    const int row0   = blockIdx.x * 16;
    const int ncol   = wave * 16 + laneLo;  // output column 0..127

    const float* bias = (mat == 0) ? bq : (mat == 1) ? bk : (mat == 2) ? bv : bs;
    float*       out  = (mat == 0) ? oq : (mat == 1) ? okk : (mat == 2) ? ov : os;

    // A: row (row0+laneLo); per-lane K chunks {0..7,16..23} (+8 for lanes 16-31)
    const _Float16* A = h16 + (size_t)(row0 + laneLo) * FDIM;
    // B[k][n] = W[n][k]: per-lane K chunks {0..7,16..23} (+8 for lanes 16-31) of row n of W
    const _Float16* B = w16 + (size_t)mat * (FDIM * FDIM) + (size_t)ncol * FDIM;

    const float bval = bias[ncol];
    v8f c;
#pragma unroll
    for (int i = 0; i < 8; ++i) c[i] = bval;

#pragma unroll
    for (int kb = 0; kb < 4; ++kb) {
        const int ka = kb * 32 + laneHi * 8;
        v8h alo = *(const v8h*)(A + ka);
        v8h ahi = *(const v8h*)(A + ka + 16);
        v16h a;
#pragma unroll
        for (int i = 0; i < 8; ++i) { a[i] = alo[i]; a[i + 8] = ahi[i]; }

        const int kbb = kb * 32 + laneHi * 8;
        v8h blo = *(const v8h*)(B + kbb);
        v8h bhi = *(const v8h*)(B + kbb + 16);
        v16h b;
#pragma unroll
        for (int i = 0; i < 8; ++i) { b[i] = blo[i]; b[i + 8] = bhi[i]; }

        c = __builtin_amdgcn_wmma_f32_16x16x32_f16(
                /*neg_a=*/false, a, /*neg_b=*/false, b,
                /*c_mod=*/(short)0, c, /*reuse_a=*/false, /*reuse_b=*/false);
    }

    // C/D layout: VGPR i -> M = i + 8*laneHi, N = laneLo
    float* O = out + (size_t)(row0 + laneHi * 8) * FDIM + ncol;
#pragma unroll
    for (int i = 0; i < 8; ++i) O[(size_t)i * FDIM] = c[i];
}

// ---------------- edge kernels (wave per edge) ----------------
__global__ __launch_bounds__(256) void k_edge_scores(
    const long long* __restrict__ srcI, const long long* __restrict__ dstI,
    const float* __restrict__ q, const float* __restrict__ kmat,
    float* __restrict__ scores, unsigned* __restrict__ menc) {
    const int e    = (int)((blockIdx.x * 256u + threadIdx.x) >> 5);
    const int lane = threadIdx.x & 31;
    if (e >= NEDGE) return;
    const int s = (int)srcI[e], d = (int)dstI[e];
    float4 qa = ((const float4*)(q    + (size_t)d * FDIM))[lane];
    float4 kb = ((const float4*)(kmat + (size_t)s * FDIM))[lane];
    float p = qa.x * kb.x + qa.y * kb.y + qa.z * kb.z + qa.w * kb.w;
#pragma unroll
    for (int m = 16; m >= 1; m >>= 1) p += __shfl_xor(p, m, 32);
    p *= 0.08838834764831845f;  // 1/sqrt(128)
    if (lane == 0) {
        scores[e] = p;
        atomicMax(menc + d, enc_f32(p));
    }
}

__global__ void k_edge_exp(const long long* __restrict__ dstI,
                           float* __restrict__ scores,
                           const unsigned* __restrict__ menc,
                           float* __restrict__ denom) {
    int e = blockIdx.x * blockDim.x + threadIdx.x;
    if (e >= NEDGE) return;
    const int d = (int)dstI[e];
    float w = __expf(scores[e] - dec_f32(menc[d]));
    scores[e] = w;
    unsafeAtomicAdd(denom + d, w);   // native global_atomic_add_f32
}

__global__ __launch_bounds__(256) void k_edge_agg(
    const long long* __restrict__ srcI, const long long* __restrict__ dstI,
    const float* __restrict__ scores, const float* __restrict__ denom,
    const float* __restrict__ v, float* __restrict__ agg) {
    const int e    = (int)((blockIdx.x * 256u + threadIdx.x) >> 5);
    const int lane = threadIdx.x & 31;
    if (e >= NEDGE) return;
    const int s = (int)srcI[e], d = (int)dstI[e];
    const float alpha = scores[e] / denom[d];
    float4 vv = ((const float4*)(v + (size_t)s * FDIM))[lane];
    float* ap = agg + (size_t)d * FDIM + lane * 4;
    unsafeAtomicAdd(ap + 0, alpha * vv.x);
    unsafeAtomicAdd(ap + 1, alpha * vv.y);
    unsafeAtomicAdd(ap + 2, alpha * vv.z);
    unsafeAtomicAdd(ap + 3, alpha * vv.w);
}

// ---------------- final FC + log_softmax (wave per node) ----------------
__global__ __launch_bounds__(256) void k_fc_logsoftmax(
    const _Float16* __restrict__ h16, const float* __restrict__ fcW,
    const float* __restrict__ fcb, float* __restrict__ out) {
    const int node = (int)((blockIdx.x * 256u + threadIdx.x) >> 5);
    const int lane = threadIdx.x & 31;
    if (node >= NNODE) return;
    float hv[4];
#pragma unroll
    for (int j = 0; j < 4; ++j)
        hv[j] = (float)h16[(size_t)node * FDIM + lane * 4 + j];
    float logits[NCLS];
#pragma unroll
    for (int c = 0; c < NCLS; ++c) {
        const float* w = fcW + c * FDIM + lane * 4;
        float p = hv[0] * w[0] + hv[1] * w[1] + hv[2] * w[2] + hv[3] * w[3];
#pragma unroll
        for (int m = 16; m >= 1; m >>= 1) p += __shfl_xor(p, m, 32);
        logits[c] = p + fcb[c];
    }
    if (lane == 0) {
        float mx = logits[0];
#pragma unroll
        for (int c = 1; c < NCLS; ++c) mx = fmaxf(mx, logits[c]);
        float se = 0.f;
#pragma unroll
        for (int c = 0; c < NCLS; ++c) se += __expf(logits[c] - mx);
        const float lse = mx + __logf(se);
#pragma unroll
        for (int c = 0; c < NCLS; ++c)
            out[(size_t)node * NCLS + c] = logits[c] - lse;
    }
}

// ---------------- host orchestration ----------------
extern "C" void kernel_launch(void* const* d_in, const int* in_sizes, int n_in,
                              void* d_out, int out_size, void* d_ws, size_t ws_size,
                              hipStream_t stream) {
    const float*     x    = (const float*)d_in[0];
    const long long* ei   = (const long long*)d_in[1];   // int64 edge_index [2,E]
    const float*     Wq   = (const float*)d_in[2];
    const float*     bq   = (const float*)d_in[3];
    const float*     Wk   = (const float*)d_in[4];
    const float*     bk   = (const float*)d_in[5];
    const float*     Wv   = (const float*)d_in[6];
    const float*     bv   = (const float*)d_in[7];
    const float*     Ws   = (const float*)d_in[8];
    const float*     bs   = (const float*)d_in[9];
    const float*     fcW  = (const float*)d_in[10];
    const float*     fcb  = (const float*)d_in[11];

    const long long* srcI = ei;          // edge_index[0]
    const long long* dstI = ei + NEDGE;  // edge_index[1]

    // workspace carve-out (256B aligned)
    char*  ws  = (char*)d_ws;
    size_t off = 0;
    auto alloc = [&](size_t bytes) -> char* {
        char* p = ws + off;
        off += (bytes + 255) & ~(size_t)255;
        return p;
    };
    _Float16* h16    = (_Float16*)alloc((size_t)NNODE * FDIM * 2);
    float*    qbuf   = (float*)alloc((size_t)NNODE * FDIM * 4);
    float*    kbuf   = (float*)alloc((size_t)NNODE * FDIM * 4);
    float*    vbuf   = (float*)alloc((size_t)NNODE * FDIM * 4);
    float*    agg    = (float*)alloc((size_t)NNODE * FDIM * 4);  // skip + attention sum
    _Float16* w16    = (_Float16*)alloc((size_t)4 * FDIM * FDIM * 2);
    float*    scores = (float*)alloc((size_t)NEDGE * 4);
    unsigned* menc   = (unsigned*)alloc((size_t)NNODE * 4);
    float*    denom  = (float*)alloc((size_t)NNODE * 4);

    const int nElem = NNODE * FDIM;                    // 6.4M, /256 exact
    k_f32_to_f16<<<(nElem + 255) / 256, 256, 0, stream>>>(x, h16, nElem);

    for (int l = 0; l < 2; ++l) {
        k_pack_weights<<<(4 * FDIM * FDIM + 255) / 256, 256, 0, stream>>>(
            Wq, Wk, Wv, Ws, w16, l * FDIM * FDIM);
        k_init_softmax<<<(NNODE + 255) / 256, 256, 0, stream>>>(menc, denom, NNODE);

        dim3 g(NNODE / 16, 4);                         // 3125 x 4
        k_gemm_qkvs<<<g, 256, 0, stream>>>(
            h16, w16, bq + l * FDIM, bk + l * FDIM, bv + l * FDIM, bs + l * FDIM,
            qbuf, kbuf, vbuf, agg);                    // skip GEMM lands in agg

        const int edgeWaveBlocks = (NEDGE * 32 + 255) / 256;   // 75000
        k_edge_scores<<<edgeWaveBlocks, 256, 0, stream>>>(srcI, dstI, qbuf, kbuf,
                                                          scores, menc);
        k_edge_exp<<<(NEDGE + 255) / 256, 256, 0, stream>>>(dstI, scores, menc, denom);
        k_edge_agg<<<edgeWaveBlocks, 256, 0, stream>>>(srcI, dstI, scores, denom,
                                                       vbuf, agg);
        k_relu_to_h16<<<(nElem + 255) / 256, 256, 0, stream>>>(agg, h16, nElem);
    }

    k_fc_logsoftmax<<<(NNODE * 32 + 255) / 256, 256, 0, stream>>>(
        h16, fcW, fcb, (float*)d_out);
}
